// LearnedWLS_2671469658616
// MI455X (gfx1250) — matrix operand order repaired and checked by reading, not compiled, not run on hardware
//
#include <hip/hip_runtime.h>
#include <hip/hip_bf16.h>

// ---------------------------------------------------------------------------
// LearnedWLS on MI455X (gfx1250): f16 WMMA MLP + per-row 4x4 WLS solve.
//
// Each wave32 processes 32 batch rows (two 16-row M-tiles). MLP layers are
// D(32xN) = A(32xK) x B(KxN) chains of v_wmma_f32_16x16x32_f16; every B
// fragment load is amortized over two WMMAs (two M-tiles), halving weight
// L2 traffic per row. Weights are pre-converted to f16 in d_ws.
// ---------------------------------------------------------------------------

typedef __attribute__((ext_vector_type(16))) _Float16 v16h;
typedef __attribute__((ext_vector_type(8)))  _Float16 v8h;
typedef __attribute__((ext_vector_type(8)))  float    v8f;

#define B_TOTAL  32768
#define M_PTS    32
#define H_DIM    64
#define NLAYERS  4
#define DIM_IN   160   // (4+1)*M_PTS

// half-element offsets inside d_ws
#define WIN16_OFF   0
#define WENC16_OFF  (64*160)
#define WOUT16_OFF  (64*160 + 4*64*64)
#define W_TOTAL_H   (64*160 + 4*64*64 + 32*64)

#define WAVES_PER_BLOCK 2
#define ROWS_PER_WAVE   32           // two 16-row WMMA M-tiles
#define ROWS_PER_BLOCK  (WAVES_PER_BLOCK * ROWS_PER_WAVE)

// ---------------------------------------------------------------------------
// Kernel 1: convert f32 weights -> f16 blob in d_ws (row-major, same shapes).
// ---------------------------------------------------------------------------
__global__ void prep_weights_f16(const float* __restrict__ W_in,
                                 const float* __restrict__ W_enc,
                                 const float* __restrict__ W_out,
                                 _Float16* __restrict__ wf16) {
    int i = blockIdx.x * blockDim.x + threadIdx.x;
    const int n0 = 64 * 160;
    const int n1 = 4 * 64 * 64;
    const int n2 = 32 * 64;
    if (i < n0) {
        wf16[WIN16_OFF + i] = (_Float16)W_in[i];
    } else if (i < n0 + n1) {
        wf16[WENC16_OFF + (i - n0)] = (_Float16)W_enc[i - n0];
    } else if (i < n0 + n1 + n2) {
        wf16[WOUT16_OFF + (i - n0 - n1)] = (_Float16)W_out[i - n0 - n1];
    }
}

// ---------------------------------------------------------------------------
// Kernel 2: MLP (WMMA) + weighted-least-squares solve.
// ---------------------------------------------------------------------------
__launch_bounds__(WAVES_PER_BLOCK * 32)
__global__ void mlp_wls_kernel(const float*         __restrict__ x,        // (B,32,4)
                               const unsigned char* __restrict__ pad_mask, // (B,32) bool
                               const float*         __restrict__ b_in,     // (64)
                               const float*         __restrict__ a_in,     // (1)
                               const float*         __restrict__ b_enc,    // (4,64)
                               const float*         __restrict__ a_enc,    // (4,1)
                               const float*         __restrict__ b_out,    // (32)
                               const _Float16*      __restrict__ wf16,     // f16 weights
                               float*               __restrict__ out)      // (B,4)
{
    __shared__ __align__(16) _Float16 feats[WAVES_PER_BLOCK][ROWS_PER_WAVE * DIM_IN];
    __shared__ __align__(16) _Float16 act  [WAVES_PER_BLOCK][2][ROWS_PER_WAVE * H_DIM];
    __shared__ __align__(16) float    wbuf [WAVES_PER_BLOCK][ROWS_PER_WAVE * M_PTS];

    const int tid  = threadIdx.x;
    const int wave = tid >> 5;
    const int lane = tid & 31;
    const int g    = lane >> 4;   // lane group (0/1) -> K sub-offset in fragments
    const int nl   = lane & 15;   // local row (A) / column (B,C,D)
    const int rowBase = blockIdx.x * ROWS_PER_BLOCK + wave * ROWS_PER_WAVE;

    // Warm the caches with the (tiny, shared) weight blob: global_prefetch_b8.
    __builtin_prefetch(wf16 + ((size_t)tid << 6), 0, 3);

    // ---- Stage features into LDS, one batch row per lane (coalesced) ----
    // feats[m][5*j+c] = x[m][j][c] (c<4), valid flag (c==4)
    {
        const int m   = lane;
        const int row = rowBase + m;
        const float4*   px4 = (const float4*)(x + (size_t)row * (M_PTS * 4));
        const unsigned* pmw = (const unsigned*)(pad_mask + (size_t)row * M_PTS);
        _Float16* f = &feats[wave][m * DIM_IN];
#pragma unroll
        for (int j = 0; j < M_PTS; ++j) {
            float4 xv = px4[j];
            unsigned w = pmw[j >> 2];
            unsigned msk = (w >> ((j & 3) * 8)) & 0xffu;
            f[5 * j + 0] = (_Float16)xv.x;
            f[5 * j + 1] = (_Float16)xv.y;
            f[5 * j + 2] = (_Float16)xv.z;
            f[5 * j + 3] = (_Float16)xv.w;
            f[5 * j + 4] = msk ? (_Float16)0.0f : (_Float16)1.0f;
        }
    }
    __syncthreads();

    // ---- Fragment loaders (ISA 7.12.2 wave32 layouts) ----
    // A (16x32 f16): lane L -> row (L&15); halves [K0+8g .. +7] then [K0+8g+16 .. +23]
    auto loadA = [&](const _Float16* base, int stride, int kc, int mt) -> v16h {
        const _Float16* p = base + (mt * 16 + nl) * stride + kc * 32 + 8 * g;
        v8h lo = *(const v8h*)(p);
        v8h hi = *(const v8h*)(p + 16);
        return __builtin_shufflevector(lo, hi, 0,1,2,3,4,5,6,7,8,9,10,11,12,13,14,15);
    };
    // B (32x16 f16): lane L -> column (L&15); halves [K0+8g .. +7] then [K0+8g+16 .. +23]
    // B[k][n] = W[n][k] so a lane reads a contiguous row segment of W (row-major).
    auto loadB = [&](const _Float16* W, int stride, int ntile, int kc) -> v16h {
        const _Float16* p = W + (ntile * 16 + nl) * stride + kc * 32 + 8 * g;
        v8h lo = *(const v8h*)(p);
        v8h hi = *(const v8h*)(p + 16);
        return __builtin_shufflevector(lo, hi, 0,1,2,3,4,5,6,7,8,9,10,11,12,13,14,15);
    };
    auto wmma = [&](v16h a, v16h b, v8f c) -> v8f {
        return __builtin_amdgcn_wmma_f32_16x16x32_f16(
            /*neg_a=*/false, a, /*neg_b=*/false, b,
            /*c_mod=*/(short)0, c, /*reuse_a=*/false, /*reuse_b=*/false);
    };

    const v8f vzero = {};
    v8f acc[4][2];   // [n-tile][m-tile]

    // ---- Input layer: (32x160) @ (160x64) ----
    {
        const _Float16* Win16 = wf16 + WIN16_OFF;
        for (int t = 0; t < 4; ++t) { acc[t][0] = vzero; acc[t][1] = vzero; }
        for (int kc = 0; kc < DIM_IN / 32; ++kc) {
            v16h a0 = loadA(&feats[wave][0], DIM_IN, kc, 0);
            v16h a1 = loadA(&feats[wave][0], DIM_IN, kc, 1);
            for (int t = 0; t < 4; ++t) {
                v16h b = loadB(Win16, DIM_IN, t, kc);
                acc[t][0] = wmma(a0, b, acc[t][0]);   // same B feeds both M-tiles
                acc[t][1] = wmma(a1, b, acc[t][1]);
            }
        }
        float alpha = a_in[0];
        for (int t = 0; t < 4; ++t) {
            int n = t * 16 + nl;
            float bias = b_in[n];
            for (int mt = 0; mt < 2; ++mt) {
                for (int v = 0; v < 8; ++v) {
                    int m = mt * 16 + v + 8 * g;     // D layout: VGPR v -> row v+8g
                    float val = acc[t][mt][v] + bias;
                    val = (val >= 0.0f) ? val : alpha * val;
                    act[wave][0][m * H_DIM + n] = (_Float16)val;
                }
            }
        }
    }
    __syncthreads();

    // ---- Encoder layers: (32x64) @ (64x64), ping-pong activation buffers ----
    int cur = 0;
    for (int L = 0; L < NLAYERS; ++L) {
        const _Float16* W = wf16 + WENC16_OFF + L * H_DIM * H_DIM;
        for (int t = 0; t < 4; ++t) { acc[t][0] = vzero; acc[t][1] = vzero; }
        for (int kc = 0; kc < H_DIM / 32; ++kc) {
            v16h a0 = loadA(&act[wave][cur][0], H_DIM, kc, 0);
            v16h a1 = loadA(&act[wave][cur][0], H_DIM, kc, 1);
            for (int t = 0; t < 4; ++t) {
                v16h b = loadB(W, H_DIM, t, kc);
                acc[t][0] = wmma(a0, b, acc[t][0]);
                acc[t][1] = wmma(a1, b, acc[t][1]);
            }
        }
        float alpha = a_enc[L];
        int nxt = cur ^ 1;
        for (int t = 0; t < 4; ++t) {
            int n = t * 16 + nl;
            float bias = b_enc[L * H_DIM + n];
            for (int mt = 0; mt < 2; ++mt) {
                for (int v = 0; v < 8; ++v) {
                    int m = mt * 16 + v + 8 * g;
                    float val = acc[t][mt][v] + bias;
                    val = (val >= 0.0f) ? val : alpha * val;
                    act[wave][nxt][m * H_DIM + n] = (_Float16)val;
                }
            }
        }
        cur = nxt;
        __syncthreads();
    }

    // ---- Output layer: (32x64) @ (64x32) -> w weights ----
    {
        const _Float16* Wout16 = wf16 + WOUT16_OFF;
        v8f oacc[2][2];
        oacc[0][0] = vzero; oacc[0][1] = vzero;
        oacc[1][0] = vzero; oacc[1][1] = vzero;
        for (int kc = 0; kc < H_DIM / 32; ++kc) {
            v16h a0 = loadA(&act[wave][cur][0], H_DIM, kc, 0);
            v16h a1 = loadA(&act[wave][cur][0], H_DIM, kc, 1);
            for (int t = 0; t < 2; ++t) {
                v16h b = loadB(Wout16, H_DIM, t, kc);
                oacc[t][0] = wmma(a0, b, oacc[t][0]);
                oacc[t][1] = wmma(a1, b, oacc[t][1]);
            }
        }
        for (int t = 0; t < 2; ++t) {
            int n = t * 16 + nl;
            float bias = b_out[n];
            for (int mt = 0; mt < 2; ++mt) {
                for (int v = 0; v < 8; ++v) {
                    int m = mt * 16 + v + 8 * g;
                    wbuf[wave][m * M_PTS + n] = oacc[t][mt][v] + bias;
                }
            }
        }
    }
    __syncthreads();

    // ---- Weighted least squares per batch row: out = (A^T W^2 A)^-1 A^T W^2 r
    // A_j = [-x1, -x2, -x3, 1], r_j = x0, weight = w_j^2. 32 lanes -> 32 rows.
    {
        const int row = rowBase + lane;
        const float* px = x + (size_t)row * (M_PTS * 4);
        float S[4][4] = {};
        float tv[4]   = {};
        for (int j = 0; j < M_PTS; ++j) {
            float wj = wbuf[wave][lane * M_PTS + j];
            float w2 = wj * wj;
            float aa[4];
            aa[0] = -px[j * 4 + 1];
            aa[1] = -px[j * 4 + 2];
            aa[2] = -px[j * 4 + 3];
            aa[3] = 1.0f;
            float r = px[j * 4 + 0];
            for (int p = 0; p < 4; ++p) {
                tv[p] += w2 * aa[p] * r;
                for (int q = 0; q < 4; ++q) S[p][q] += w2 * aa[p] * aa[q];
            }
        }
        // 4x4 Gaussian elimination (SPD normal equations; no pivoting)
        for (int k = 0; k < 4; ++k) {
            float inv = 1.0f / S[k][k];
            for (int i = k + 1; i < 4; ++i) {
                float f = S[i][k] * inv;
                for (int q = k; q < 4; ++q) S[i][q] -= f * S[k][q];
                tv[i] -= f * tv[k];
            }
        }
        float y[4];
        for (int k = 3; k >= 0; --k) {
            float s = tv[k];
            for (int q = k + 1; q < 4; ++q) s -= S[k][q] * y[q];
            y[k] = s / S[k][k];
        }
        for (int c = 0; c < 4; ++c) out[(size_t)row * 4 + c] = y[c];
    }
}

// ---------------------------------------------------------------------------
extern "C" void kernel_launch(void* const* d_in, const int* in_sizes, int n_in,
                              void* d_out, int out_size, void* d_ws, size_t ws_size,
                              hipStream_t stream) {
    (void)in_sizes; (void)n_in; (void)out_size; (void)ws_size;
    const float*         x     = (const float*)d_in[0];
    const unsigned char* pmask = (const unsigned char*)d_in[1]; // jax bool -> 1 byte
    const float*         W_in  = (const float*)d_in[2];
    const float*         b_in  = (const float*)d_in[3];
    const float*         a_in  = (const float*)d_in[4];
    const float*         W_enc = (const float*)d_in[5];
    const float*         b_enc = (const float*)d_in[6];
    const float*         a_enc = (const float*)d_in[7];
    const float*         W_out = (const float*)d_in[8];
    const float*         b_out = (const float*)d_in[9];
    float*               outp  = (float*)d_out;
    _Float16*            wf16  = (_Float16*)d_ws;

    const int totalW = W_TOTAL_H;
    prep_weights_f16<<<(totalW + 255) / 256, 256, 0, stream>>>(W_in, W_enc, W_out, wf16);

    const int blocks = B_TOTAL / ROWS_PER_BLOCK; // 32768 / 64 = 512
    mlp_wls_kernel<<<blocks, WAVES_PER_BLOCK * 32, 0, stream>>>(
        x, pmask, b_in, a_in, b_enc, a_enc, b_out, wf16, outp);
}
